// AVPViT_91087666413999
// MI455X (gfx1250) — matrix operand order repaired and checked by reading, not compiled, not run on hardware
//
#include <hip/hip_runtime.h>
#include <hip/hip_bf16.h>

// ---------------- model constants ----------------
#define BSZ     16
#define NVIEW   2
#define IMGSZ   224
#define GRID    7
#define PATCH   14
#define SGRID   16
#define DMODEL  768
#define NHEADS  12
#define NLAYER  12
#define DHEAD   64
#define NLOC    (GRID*GRID)     // 49
#define NSCN    (SGRID*SGRID)   // 256
#define GSIZE   (GRID*PATCH)    // 98

typedef __attribute__((ext_vector_type(4)))  float  f32x4;
typedef __attribute__((ext_vector_type(8)))  float  f32x8;
typedef __attribute__((ext_vector_type(8)))  __bf16 bf16x8;
typedef __attribute__((ext_vector_type(16))) __bf16 bf16x16;

union FragAB { bf16x8 part[2]; bf16x16 whole; };

__device__ __forceinline__ float gelu_tanh(float x) {
  float x3 = x * x * x;
  return 0.5f * x * (1.f + tanhf(0.7978845608028654f * (x + 0.044715f * x3)));
}

// =====================================================================
// WMMA bf16 GEMM:  Y[M,N] = X[M,K] * W[N,K]^T + bias, optional GELU.
// Block = 256 threads (8 wave32s), tile 64x128, K-step 32.
// Software-pipelined double-buffered LDS: global loads for chunk i+1 are
// issued BEFORE the WMMAs of chunk i, so HBM latency overlaps the matrix
// pipe (the kernel is weight-bandwidth bound: ~1.1 GB fp32 weights/call
// vs us-scale WMMA math). fp32 -> bf16 conversion happens at LDS store.
// =====================================================================
#define BM 64
#define BN 128
#define BK 32
#define LDSTR 40   // bf16 elements per LDS row (80B) -> conflict-free b128 frag loads

__global__ __launch_bounds__(256)
void k_gemm_bf16(const float* __restrict__ X, const float* __restrict__ W,
                 const float* __restrict__ bias, float* __restrict__ Y,
                 int M, int N, int K, int act)
{
  __shared__ __align__(16) __bf16 sA[2][BM * LDSTR];
  __shared__ __align__(16) __bf16 sB[2][BN * LDSTR];

  const int tid  = threadIdx.x;
  const int bm   = blockIdx.y, bn = blockIdx.x;
  const int w    = tid >> 5,  lane = tid & 31;
  const int l    = lane & 15, hi   = lane >> 4;     // half-wave
  const int wm   = w & 3,     wn   = w >> 2;        // 4x2 wave grid

  // staging maps: A tile 64x32 (8 f32/thread), B tile 128x32 (16 f32/thread)
  const int arow = tid >> 2, akseg = (tid & 3) * 8;
  const int grow = bm * BM + arow;
  const int brow = tid >> 1, bkseg = (tid & 1) * 16;
  const int gn   = bn * BN + brow;
  const bool aok = grow < M;
  const bool bok = gn < N;

  f32x8 acc[4];
#pragma unroll
  for (int c = 0; c < 4; ++c) {
    const int col = bn * BN + wn * 64 + c * 16 + l;
    const float bv = (bias != nullptr && col < N) ? bias[col] : 0.f;
#pragma unroll
    for (int r = 0; r < 8; ++r) acc[c][r] = bv;
  }

  const int nk = K / BK;
  f32x4 a0, a1, b0, b1, b2, b3;

  auto load_chunk = [&](int kc) {
    a0 = f32x4{0.f,0.f,0.f,0.f}; a1 = a0;
    b0 = a0; b1 = a0; b2 = a0; b3 = a0;
    if (aok) {
      const float* g = X + (size_t)grow * K + kc + akseg;
      a0 = *(const f32x4*)g;
      a1 = *(const f32x4*)(g + 4);
      if (kc + BK < K) __builtin_prefetch(g + BK, 0, 1);   // global_prefetch_b8
    }
    if (bok) {
      const float* g = W + (size_t)gn * K + kc + bkseg;
      b0 = *(const f32x4*)g;       b1 = *(const f32x4*)(g + 4);
      b2 = *(const f32x4*)(g + 8); b3 = *(const f32x4*)(g + 12);
      if (kc + BK < K) __builtin_prefetch(g + BK, 0, 1);
    }
  };
  auto store_chunk = [&](int buf) {
    bf16x8 h;
    h[0]=(__bf16)a0[0]; h[1]=(__bf16)a0[1]; h[2]=(__bf16)a0[2]; h[3]=(__bf16)a0[3];
    h[4]=(__bf16)a1[0]; h[5]=(__bf16)a1[1]; h[6]=(__bf16)a1[2]; h[7]=(__bf16)a1[3];
    *(bf16x8*)(&sA[buf][arow * LDSTR + akseg]) = h;
    bf16x8 h0, h1;
    h0[0]=(__bf16)b0[0]; h0[1]=(__bf16)b0[1]; h0[2]=(__bf16)b0[2]; h0[3]=(__bf16)b0[3];
    h0[4]=(__bf16)b1[0]; h0[5]=(__bf16)b1[1]; h0[6]=(__bf16)b1[2]; h0[7]=(__bf16)b1[3];
    h1[0]=(__bf16)b2[0]; h1[1]=(__bf16)b2[1]; h1[2]=(__bf16)b2[2]; h1[3]=(__bf16)b2[3];
    h1[4]=(__bf16)b3[0]; h1[5]=(__bf16)b3[1]; h1[6]=(__bf16)b3[2]; h1[7]=(__bf16)b3[3];
    *(bf16x8*)(&sB[buf][brow * LDSTR + bkseg])     = h0;
    *(bf16x8*)(&sB[buf][brow * LDSTR + bkseg + 8]) = h1;
  };

  // prologue: stage chunk 0 into buffer 0
  load_chunk(0);
  store_chunk(0);
  __syncthreads();

  for (int i = 0; i < nk; ++i) {
    const int cur = i & 1;
    const bool more = (i + 1) < nk;

    // issue next chunk's global loads first: they fly while WMMAs run
    if (more) load_chunk((i + 1) * BK);

    // A fragment: 16-bit A layout (lo lanes K 0-7 & 16-23, hi lanes K 8-15 & 24-31)
    FragAB fa;
    const __bf16* pa = &sA[cur][(wm * 16 + l) * LDSTR + hi * 8];
    fa.part[0] = *(const bf16x8*)pa;
    fa.part[1] = *(const bf16x8*)(pa + 16);
#pragma unroll
    for (int c = 0; c < 4; ++c) {
      // B fragment: lane = N, lo lanes K 0-7 & 16-23, hi lanes K 8-15 & 24-31
      FragAB fb;
      const __bf16* pb = &sB[cur][(wn * 64 + c * 16 + l) * LDSTR + hi * 8];
      fb.part[0] = *(const bf16x8*)pb;
      fb.part[1] = *(const bf16x8*)(pb + 16);
      acc[c] = __builtin_amdgcn_wmma_f32_16x16x32_bf16(
          false, fa.whole, false, fb.whole, (short)0, acc[c], false, false);
    }

    // convert + park next chunk into the other buffer, then one barrier
    if (more) store_chunk(cur ^ 1);
    __syncthreads();
  }

#pragma unroll
  for (int c = 0; c < 4; ++c) {
    const int col = bn * BN + wn * 64 + c * 16 + l;
    if (col >= N) continue;
#pragma unroll
    for (int r = 0; r < 8; ++r) {
      const int row = bm * BM + wm * 16 + hi * 8 + r;
      if (row < M) {
        float v = acc[c][r];
        if (act == 1) v = gelu_tanh(v);
        Y[(size_t)row * N + col] = v;
      }
    }
  }
}

// =====================================================================
// Flash-style attention: one query per lane, online softmax, K/V tiles
// staged through LDS. O is written merged: [b*Tq+qi, 768] at col h*64.
// =====================================================================
__global__ __launch_bounds__(64)
void k_attn(const float* __restrict__ Q, int ldq,
            const float* __restrict__ Kb, int ldk,
            const float* __restrict__ Vb, int ldv,
            float* __restrict__ O, int Tq, int Tk, float scale)
{
  __shared__ float sK[32][DHEAD];
  __shared__ float sV[32][DHEAD];
  const int tid = threadIdx.x;
  const int bh  = blockIdx.x;
  const int b   = bh / NHEADS, h = bh % NHEADS;
  const int qi  = blockIdx.y * 64 + tid;
  const bool active = qi < Tq;

  float q[DHEAD];
  if (active) {
    const float* qp = Q + (size_t)(b * Tq + qi) * ldq + h * DHEAD;
#pragma unroll
    for (int d = 0; d < DHEAD; ++d) q[d] = qp[d];
  }
  float acc[DHEAD];
#pragma unroll
  for (int d = 0; d < DHEAD; ++d) acc[d] = 0.f;
  float m = -3.0e38f, lsum = 0.f;

  for (int kt = 0; kt < Tk; kt += 32) {
    const int jmax = min(32, Tk - kt);
    for (int i = tid; i < jmax * DHEAD; i += 64) {
      const int j = i >> 6, d = i & 63;
      sK[j][d] = Kb[(size_t)(b * Tk + kt + j) * ldk + h * DHEAD + d];
      sV[j][d] = Vb[(size_t)(b * Tk + kt + j) * ldv + h * DHEAD + d];
    }
    __syncthreads();
    if (active) {
      for (int j = 0; j < jmax; ++j) {
        float s = 0.f;
#pragma unroll
        for (int d = 0; d < DHEAD; ++d) s += q[d] * sK[j][d];
        s *= scale;
        const float nm  = fmaxf(m, s);
        const float f   = __expf(m - nm);
        const float wgt = __expf(s - nm);
        lsum = lsum * f + wgt;
#pragma unroll
        for (int d = 0; d < DHEAD; ++d) acc[d] = acc[d] * f + wgt * sV[j][d];
        m = nm;
      }
    }
    __syncthreads();
  }
  if (active) {
    const float inv = 1.f / lsum;
    float* op = O + (size_t)(b * Tq + qi) * DMODEL + h * DHEAD;
#pragma unroll
    for (int d = 0; d < DHEAD; ++d) op[d] = acc[d] * inv;
  }
}

// ----------------------------- LayerNorm -----------------------------
__global__ __launch_bounds__(256)
void k_ln(const float* __restrict__ X, const float* __restrict__ w,
          const float* __restrict__ bv, float* __restrict__ Y)
{
  __shared__ float red[256];
  const int tid = threadIdx.x;
  const int row = blockIdx.x;
  const float* x = X + (size_t)row * DMODEL;
  float s = 0.f;
  for (int d = tid; d < DMODEL; d += 256) s += x[d];
  red[tid] = s; __syncthreads();
  for (int o = 128; o > 0; o >>= 1) { if (tid < o) red[tid] += red[tid + o]; __syncthreads(); }
  const float mean = red[0] * (1.f / DMODEL);
  __syncthreads();
  float v = 0.f;
  for (int d = tid; d < DMODEL; d += 256) { float t = x[d] - mean; v += t * t; }
  red[tid] = v; __syncthreads();
  for (int o = 128; o > 0; o >>= 1) { if (tid < o) red[tid] += red[tid + o]; __syncthreads(); }
  const float rstd = rsqrtf(red[0] * (1.f / DMODEL) + 1e-6f);
  for (int d = tid; d < DMODEL; d += 256)
    Y[(size_t)row * DMODEL + d] = (x[d] - mean) * rstd * w[d] + bv[d];
}

// --------------------------- elementwise -----------------------------
__global__ void k_scene_init(const float* __restrict__ tok, float* __restrict__ scene, int n) {
  int i = blockIdx.x * blockDim.x + threadIdx.x;
  if (i < n) scene[i] = tok[i % (NSCN * DMODEL)];
}
__global__ void k_add(float* __restrict__ x, const float* __restrict__ t, int n) {
  int i = blockIdx.x * blockDim.x + threadIdx.x;
  if (i < n) x[i] += t[i];
}
__global__ void k_addgate(float* __restrict__ x, const float* __restrict__ t,
                          const float* __restrict__ gate, int n) {
  int i = blockIdx.x * blockDim.x + threadIdx.x;
  if (i < n) x[i] += gate[i % DMODEL] * t[i];
}
__global__ void k_copy(float* __restrict__ dst, const float* __restrict__ src, int n) {
  int i = blockIdx.x * blockDim.x + threadIdx.x;
  if (i < n) dst[i] = src[i];
}

// --------------------------- RoPE tables -----------------------------
__global__ void k_rope_scene(float* __restrict__ cs, float* __restrict__ sn) {
  int i = blockIdx.x * blockDim.x + threadIdx.x;
  if (i >= NSCN * 32) return;
  int s = i >> 5, j = i & 31;
  int coord = j >> 4, p = j & 15;
  int r = s >> 4, c = s & 15;
  float x = -1.f + 2.f * c / 15.f;
  float y = -1.f + 2.f * r / 15.f;
  float pos = coord ? y : x;
  float ang = pos / powf(100.f, (float)p / 16.f);
  cs[i] = cosf(ang); sn[i] = sinf(ang);
}
__global__ void k_rope_local(const float* __restrict__ centers, const float* __restrict__ scales,
                             int vp, float* __restrict__ cs, float* __restrict__ sn) {
  int i = blockIdx.x * blockDim.x + threadIdx.x;
  if (i >= BSZ * NLOC * 32) return;
  int b = i / (NLOC * 32);
  int rem = i % (NLOC * 32);
  int nn = rem >> 5, j = rem & 31;
  int coord = j >> 4, p = j & 15;
  int r = nn / GRID, c = nn % GRID;
  float gx = -1.f + 2.f * c / 6.f;
  float gy = -1.f + 2.f * r / 6.f;
  float cx = centers[(vp * BSZ + b) * 2 + 0];
  float cy = centers[(vp * BSZ + b) * 2 + 1];
  float sc = scales[vp * BSZ + b];
  float pos = coord ? (cy + sc * gy) : (cx + sc * gx);
  float ang = pos / powf(100.f, (float)p / 16.f);
  cs[i] = cosf(ang); sn[i] = sinf(ang);
}
// Apply rope in-place to all 12 heads of a [B*T, ld] buffer (rows start at col 0).
__global__ void k_rope_apply(float* __restrict__ x, int ld, int T,
                             const float* __restrict__ cs, const float* __restrict__ sn,
                             int perBatch, int n) {
  int i = blockIdx.x * blockDim.x + threadIdx.x;
  if (i >= n) return;
  int row = i / (NHEADS * 32);
  int rem = i % (NHEADS * 32);
  int h = rem >> 5, j = rem & 31;
  int t = row % T;
  int cidx = (perBatch ? row : t) * 32 + j;
  float c = cs[cidx], s = sn[cidx];
  float* p = x + (size_t)row * ld + h * DHEAD + 2 * j;
  float x1 = p[0], x2 = p[1];
  p[0] = x1 * c - x2 * s;
  p[1] = x1 * s + x2 * c;
}

// ------------------- glimpse extraction + patch conv ------------------
__global__ void k_glimpse(const float* __restrict__ img, const float* __restrict__ centers,
                          const float* __restrict__ scales, int vp, float* __restrict__ out, int n) {
  int i = blockIdx.x * blockDim.x + threadIdx.x;
  if (i >= n) return;
  int b = i / (3 * GSIZE * GSIZE);
  int rem = i % (3 * GSIZE * GSIZE);
  int ch = rem / (GSIZE * GSIZE);
  int ij = rem % (GSIZE * GSIZE);
  int iy = ij / GSIZE, ix = ij % GSIZE;
  float cx = centers[(vp * BSZ + b) * 2 + 0];
  float cy = centers[(vp * BSZ + b) * 2 + 1];
  float sc = scales[vp * BSZ + b];
  float gy = cy + sc * (-1.f + 2.f * iy / (GSIZE - 1.f));
  float gx = cx + sc * (-1.f + 2.f * ix / (GSIZE - 1.f));
  float py = (gy + 1.f) * 0.5f * (IMGSZ - 1.f);
  float px = (gx + 1.f) * 0.5f * (IMGSZ - 1.f);
  float y0f = fminf(fmaxf(floorf(py), 0.f), (float)(IMGSZ - 2));
  float x0f = fminf(fmaxf(floorf(px), 0.f), (float)(IMGSZ - 2));
  int y0 = (int)y0f, x0 = (int)x0f;
  float wy = fminf(fmaxf(py - y0f, 0.f), 1.f);
  float wx = fminf(fmaxf(px - x0f, 0.f), 1.f);
  const float* p = img + (((size_t)b * 3 + ch) * IMGSZ + y0) * IMGSZ + x0;
  float v00 = p[0], v01 = p[1], v10 = p[IMGSZ], v11 = p[IMGSZ + 1];
  out[i] = (v00 * (1.f - wy) + v10 * wy) * (1.f - wx) + (v01 * (1.f - wy) + v11 * wy) * wx;
}

__global__ void k_patch(const float* __restrict__ g, const float* __restrict__ pw,
                        const float* __restrict__ pb, float* __restrict__ local, int n) {
  int i = blockIdx.x * blockDim.x + threadIdx.x;
  if (i >= n) return;
  int b = i / (NLOC * DMODEL);
  int rem = i % (NLOC * DMODEL);
  int nn = rem / DMODEL, d = rem % DMODEL;
  int py = nn / GRID, px = nn % GRID;
  float s = pb[d];
  for (int c = 0; c < 3; ++c)
    for (int ky = 0; ky < PATCH; ++ky) {
      const float* gp = g + (((size_t)b * 3 + c) * GSIZE + py * PATCH + ky) * GSIZE + px * PATCH;
      const float* wp = pw + (((size_t)d * 3 + c) * PATCH + ky) * PATCH;
#pragma unroll
      for (int kx = 0; kx < PATCH; ++kx) s += gp[kx] * wp[kx];
    }
  local[i] = s;
}

// =====================================================================
// Host orchestration
// =====================================================================
extern "C" void kernel_launch(void* const* d_in, const int* in_sizes, int n_in,
                              void* d_out, int out_size, void* d_ws, size_t ws_size,
                              hipStream_t stream) {
  (void)in_sizes; (void)n_in; (void)ws_size;
  const float* images  = (const float*)d_in[0];
  const float* centers = (const float*)d_in[1];
  const float* scales  = (const float*)d_in[2];
  const float* patch_w = (const float*)d_in[3];
  const float* patch_b = (const float*)d_in[4];
  const float* stok    = (const float*)d_in[5];
  const float* ln1w = (const float*)d_in[6];
  const float* ln1b = (const float*)d_in[7];
  const float* ln2w = (const float*)d_in[8];
  const float* ln2b = (const float*)d_in[9];
  const float* qkvw = (const float*)d_in[10];
  const float* qkvb = (const float*)d_in[11];
  const float* aow  = (const float*)d_in[12];
  const float* aob  = (const float*)d_in[13];
  const float* m1w  = (const float*)d_in[14];
  const float* m1b  = (const float*)d_in[15];
  const float* m2w  = (const float*)d_in[16];
  const float* m2b  = (const float*)d_in[17];
  const float* rqw  = (const float*)d_in[18];
  const float* rqb  = (const float*)d_in[19];
  const float* rkvw = (const float*)d_in[20];
  const float* rkvb = (const float*)d_in[21];
  const float* row_ = (const float*)d_in[22];
  const float* rob  = (const float*)d_in[23];
  const float* rg   = (const float*)d_in[24];
  const float* wqw  = (const float*)d_in[25];
  const float* wqb  = (const float*)d_in[26];
  const float* wkvw = (const float*)d_in[27];
  const float* wkvb = (const float*)d_in[28];
  const float* wow  = (const float*)d_in[29];
  const float* wob  = (const float*)d_in[30];
  const float* wg   = (const float*)d_in[31];

  // ---- workspace layout (floats) : ~99 MB total ----
  float* ws      = (float*)d_ws;
  float* scene   = ws;                    // 16*256*768  = 3,145,728
  float* local   = scene   + 3145728;     // 16*49*768   =   602,112
  float* glimpse = local   + 602112;      // 16*3*98*98  =   460,992
  float* lcos    = glimpse + 460992;      // 16*49*32    =    25,088
  float* lsin    = lcos    + 25088;
  float* scos    = lsin    + 25088;       // 256*32      =     8,192
  float* ssin    = scos    + 8192;
  float* qbuf    = ssin    + 8192;        // 4096*768    = 3,145,728
  float* kvbuf   = qbuf    + 3145728;     // 4096*1536   = 6,291,456
  float* qkvbuf  = kvbuf   + 6291456;     // 784*2304    = 1,806,336
  float* attnout = qkvbuf  + 1806336;     // 4096*768    = 3,145,728
  float* tmpb    = attnout + 3145728;     // 4096*768    = 3,145,728
  float* hbuf    = tmpb    + 3145728;     // 784*768     =   602,112
  float* mbuf    = hbuf    + 602112;      // 784*3072    = 2,408,448

  const int ML = BSZ * NLOC;   // 784
  const int MS = BSZ * NSCN;   // 4096

  auto gemm = [&](const float* X, const float* Wt, const float* Bv, float* Yv,
                  int M, int N, int K, int act) {
    dim3 g((N + BN - 1) / BN, (M + BM - 1) / BM);
    k_gemm_bf16<<<g, 256, 0, stream>>>(X, Wt, Bv, Yv, M, N, K, act);
  };
  auto rope = [&](float* x, int ld, int T, const float* cs, const float* sn, int perBatch) {
    int n = BSZ * T * NHEADS * 32;
    k_rope_apply<<<(n + 255) / 256, 256, 0, stream>>>(x, ld, T, cs, sn, perBatch, n);
  };
  auto attn = [&](const float* Qp, int ldq, const float* Kp, int ldk,
                  const float* Vp, int ldv, float* Op, int Tq, int Tk) {
    dim3 g(BSZ * NHEADS, (Tq + 63) / 64);
    k_attn<<<g, 64, 0, stream>>>(Qp, ldq, Kp, ldk, Vp, ldv, Op, Tq, Tk, 0.125f);
  };

  // ---- static init ----
  { int n = MS * DMODEL; k_scene_init<<<(n + 255) / 256, 256, 0, stream>>>(stok, scene, n); }
  { int n = NSCN * 32;   k_rope_scene<<<(n + 255) / 256, 256, 0, stream>>>(scos, ssin); }

  for (int vp = 0; vp < NVIEW; ++vp) {
    { int n = BSZ * 3 * GSIZE * GSIZE;
      k_glimpse<<<(n + 255) / 256, 256, 0, stream>>>(images, centers, scales, vp, glimpse, n); }
    { int n = ML * DMODEL;
      k_patch<<<(n + 255) / 256, 256, 0, stream>>>(glimpse, patch_w, patch_b, local, n); }
    { int n = BSZ * NLOC * 32;
      k_rope_local<<<(n + 255) / 256, 256, 0, stream>>>(centers, scales, vp, lcos, lsin); }

    for (int i = 0; i < NLAYER; ++i) {
      // ---------------- READ cross-attn: local <- scene ----------------
      gemm(local, rqw  + (size_t)i * DMODEL * DMODEL, rqb  + (size_t)i * DMODEL,
           qbuf, ML, DMODEL, DMODEL, 0);
      gemm(scene, rkvw + (size_t)i * 2 * DMODEL * DMODEL, rkvb + (size_t)i * 2 * DMODEL,
           kvbuf, MS, 2 * DMODEL, DMODEL, 0);
      rope(qbuf, DMODEL, NLOC, lcos, lsin, 1);
      rope(kvbuf, 2 * DMODEL, NSCN, scos, ssin, 0);
      attn(qbuf, DMODEL, kvbuf, 2 * DMODEL, kvbuf + DMODEL, 2 * DMODEL, attnout, NLOC, NSCN);
      gemm(attnout, row_ + (size_t)i * DMODEL * DMODEL, rob + (size_t)i * DMODEL,
           tmpb, ML, DMODEL, DMODEL, 0);
      { int n = ML * DMODEL;
        k_addgate<<<(n + 255) / 256, 256, 0, stream>>>(local, tmpb, rg + (size_t)i * DMODEL, n); }

      // ---------------- local self-attention ----------------
      k_ln<<<ML, 256, 0, stream>>>(local, ln1w + (size_t)i * DMODEL, ln1b + (size_t)i * DMODEL, hbuf);
      gemm(hbuf, qkvw + (size_t)i * 3 * DMODEL * DMODEL, qkvb + (size_t)i * 3 * DMODEL,
           qkvbuf, ML, 3 * DMODEL, DMODEL, 0);
      rope(qkvbuf,          3 * DMODEL, NLOC, lcos, lsin, 1);   // q
      rope(qkvbuf + DMODEL, 3 * DMODEL, NLOC, lcos, lsin, 1);   // k
      attn(qkvbuf, 3 * DMODEL, qkvbuf + DMODEL, 3 * DMODEL,
           qkvbuf + 2 * DMODEL, 3 * DMODEL, attnout, NLOC, NLOC);
      gemm(attnout, aow + (size_t)i * DMODEL * DMODEL, aob + (size_t)i * DMODEL,
           tmpb, ML, DMODEL, DMODEL, 0);
      { int n = ML * DMODEL; k_add<<<(n + 255) / 256, 256, 0, stream>>>(local, tmpb, n); }

      // ---------------- MLP ----------------
      k_ln<<<ML, 256, 0, stream>>>(local, ln2w + (size_t)i * DMODEL, ln2b + (size_t)i * DMODEL, hbuf);
      gemm(hbuf, m1w + (size_t)i * 4 * DMODEL * DMODEL, m1b + (size_t)i * 4 * DMODEL,
           mbuf, ML, 4 * DMODEL, DMODEL, 1 /*gelu*/);
      gemm(mbuf, m2w + (size_t)i * 4 * DMODEL * DMODEL, m2b + (size_t)i * DMODEL,
           tmpb, ML, DMODEL, 4 * DMODEL, 0);
      { int n = ML * DMODEL; k_add<<<(n + 255) / 256, 256, 0, stream>>>(local, tmpb, n); }

      // ---------------- WRITE cross-attn: scene <- local ----------------
      gemm(scene, wqw  + (size_t)i * DMODEL * DMODEL, wqb  + (size_t)i * DMODEL,
           qbuf, MS, DMODEL, DMODEL, 0);
      gemm(local, wkvw + (size_t)i * 2 * DMODEL * DMODEL, wkvb + (size_t)i * 2 * DMODEL,
           kvbuf, ML, 2 * DMODEL, DMODEL, 0);
      rope(qbuf, DMODEL, NSCN, scos, ssin, 0);
      rope(kvbuf, 2 * DMODEL, NLOC, lcos, lsin, 1);
      attn(qbuf, DMODEL, kvbuf, 2 * DMODEL, kvbuf + DMODEL, 2 * DMODEL, attnout, NSCN, NLOC);
      gemm(attnout, wow + (size_t)i * DMODEL * DMODEL, wob + (size_t)i * DMODEL,
           tmpb, MS, DMODEL, DMODEL, 0);
      { int n = MS * DMODEL;
        k_addgate<<<(n + 255) / 256, 256, 0, stream>>>(scene, tmpb, wg + (size_t)i * DMODEL, n); }
    }
  }

  k_copy<<<(out_size + 255) / 256, 256, 0, stream>>>((float*)d_out, scene, out_size);
}
